// NlDecoder_79396765433883
// MI455X (gfx1250) — matrix operand order repaired and checked by reading, not compiled, not run on hardware
//
#include <hip/hip_runtime.h>
#include <hip/hip_bf16.h>

// ---------------------------------------------------------------------------
// Problem constants (B=32, S=64, H=512, V=32000, T=64)
// ---------------------------------------------------------------------------
#define Bc 32
#define Sc 64
#define Hc 512
#define Vc 32000
#define Tc 64

typedef __attribute__((ext_vector_type(16))) __bf16   v16bf;
typedef __attribute__((ext_vector_type(8)))  float    v8f;
typedef __attribute__((ext_vector_type(4)))  unsigned uint4v;
typedef __attribute__((ext_vector_type(4)))  float    float4v;

union BF16x16 {
  v16bf    v;
  uint4v   q[2];
  unsigned u[8];
  __bf16   h[16];
};

// ---------------------------------------------------------------------------
// WMMA fragment loaders, following CDNA5 ISA VGPR layouts (CDNA5 ISA)
//  A 16x32 bf16 : lane<16 -> row=lane; per lane the fragment is two contiguous
//                 8-element (16 B) runs: K = k0..k0+7 and k0+16..k0+23 with
//                 k0 = kbase + 8*(lane>>4)  ->  two ds_load_b128 per fragment.
//  B 32x16 bf16 : col=lane&15, same K order as A (k0..k0+7, k0+16..k0+23).
//  C/D 16x16 f32: M = elem + 8*(lane>>4), N = lane&15
// ---------------------------------------------------------------------------
__device__ __forceinline__ v16bf aFragLDSbf16(const __bf16* src, int row16, int ld,
                                              int kbase, int lane) {
  const int half = lane >> 4, r = row16 + (lane & 15);
  const int k0 = kbase + (half << 3);
  const uint4v* p = (const uint4v*)(src + r * ld + k0);  // 16B-aligned (ld%8==0)
  BF16x16 u;
  u.q[0] = p[0];  // K = k0 .. k0+7
  u.q[1] = p[2];  // K = k0+16 .. k0+23
  return u.v;
}

__device__ __forceinline__ v16bf aFragLDSf32(const float* src, int row16, int ld,
                                             int kbase, int lane) {
  const int half = lane >> 4, r = row16 + (lane & 15);
  const int k0 = kbase + (half << 3);
  const float4v* p = (const float4v*)(src + r * ld + k0);
  const float4v f0 = p[0], f1 = p[1];  // K = k0 .. k0+7
  const float4v f2 = p[4], f3 = p[5];  // K = k0+16 .. k0+23
  BF16x16 u;
#pragma unroll
  for (int i = 0; i < 4; ++i) {
    u.h[i]      = (__bf16)f0[i];
    u.h[4 + i]  = (__bf16)f1[i];
    u.h[8 + i]  = (__bf16)f2[i];
    u.h[12 + i] = (__bf16)f3[i];
  }
  return u.v;
}

__device__ __forceinline__ v16bf bFragG(const __bf16* __restrict__ W, int n0, int ld,
                                        int kbase, int lane) {
  const int col = lane & 15, k0 = kbase + ((lane >> 4) << 3);
  const uint4v* p = (const uint4v*)(W + (size_t)(n0 + col) * ld + k0);
  BF16x16 u;
  u.q[0] = p[0];
  u.q[1] = p[2];
  return u.v;
}

// ---------------------------------------------------------------------------
// fp32 -> bf16 conversion (one-time weight/activation down-conversion)
// ---------------------------------------------------------------------------
__global__ void cvt_f32_bf16(const float* __restrict__ s, __bf16* __restrict__ d, int n) {
  const int i = blockIdx.x * 256 + threadIdx.x;
  if (i < n) d[i] = (__bf16)s[i];
}

// ---------------------------------------------------------------------------
// Teacher-forced token embedding gather for all 64 steps:
//   tok(t,b) = 0 (SOS) if t==0 else target[b, t-1]
//   out[(t*B + b)*H + k] = bf16(emb[tok][k])
// ---------------------------------------------------------------------------
__global__ void gather_emb(const float* __restrict__ emb, const int* __restrict__ tgt,
                           __bf16* __restrict__ out) {
  const int idx = blockIdx.x * 256 + threadIdx.x;  // exactly T*B*H threads launched
  const int k  = idx & (Hc - 1);
  const int tb = idx >> 9;
  const int t = tb >> 5, b = tb & 31;
  const int tok = (t == 0) ? 0 : tgt[b * Tc + (t - 1)];
  out[idx] = (__bf16)emb[(size_t)tok * Hc + k];
}

// ---------------------------------------------------------------------------
// Generic WMMA GEMM:  out[m, n] = sum_k A[m,k] * W[n,k] + bias[n]
//   A bf16 [M,K] row-major (M = 32*gridDim.y), W bf16 [N,K] row-major, K <= 512.
//   Block = 256 threads (8 waves). Block tile = 32 (M) x 128 (N); the 32xK A
//   slab lives in LDS (32 KB) shared by all 8 waves; each wave owns one 16-col
//   N-subtile and carries TWO accumulators (M rows 0-15 and 16-31) so every
//   B fragment feeds two v_wmma_f32_16x16x32_bf16 ops (2x B reuse).
//   permuteTB: remap output row m = t*B + b  ->  b*T + t   (for [T,B]->[B,T]).
// ---------------------------------------------------------------------------
__global__ void wmma_gemm_bias(const __bf16* __restrict__ A, const __bf16* __restrict__ W,
                               const float* __restrict__ bias, float* __restrict__ out,
                               int N, int K, int permuteTB) {
  __shared__ __bf16 As[32 * 512];
  const int tid  = threadIdx.x;
  const int m0   = blockIdx.y << 5;
  const int nBlk = blockIdx.x << 7;

  // Cooperative load of the 32xK A slab (contiguous rows) into LDS.
  {
    const uint4v* src = (const uint4v*)(A + (size_t)m0 * K);
    uint4v* dst = (uint4v*)As;
    const int nvec = (32 * K) >> 3;  // 16B vectors
    for (int i = tid; i < nvec; i += 256) dst[i] = src[i];
  }
  __syncthreads();

  const int w = tid >> 5, lane = tid & 31, col = lane & 15, half = lane >> 4;
  const int n0 = nBlk + (w << 4);
  const float bi = bias[n0 + col];
  v8f acc0, acc1;
#pragma unroll
  for (int v = 0; v < 8; ++v) { acc0[v] = bi; acc1[v] = bi; }

  const int nks = K >> 5;
  for (int ks = 0; ks < nks; ++ks) {
    __builtin_prefetch(W + (size_t)(n0 + col) * K + (ks << 5) + 128, 0, 1);
    v16bf bb = bFragG(W, n0, K, ks << 5, lane);
    v16bf a0 = aFragLDSbf16(As, 0, K, ks << 5, lane);
    v16bf a1 = aFragLDSbf16(As, 16, K, ks << 5, lane);
    acc0 = __builtin_amdgcn_wmma_f32_16x16x32_bf16(false, a0, false, bb, (short)0, acc0,
                                                   false, false);
    acc1 = __builtin_amdgcn_wmma_f32_16x16x32_bf16(false, a1, false, bb, (short)0, acc1,
                                                   false, false);
  }

#pragma unroll
  for (int v = 0; v < 8; ++v) {
    const int mA = m0 + v + (half << 3);
    const int mB = mA + 16;
    const int rA = permuteTB ? ((mA & (Bc - 1)) * Tc + (mA >> 5)) : mA;
    const int rB = permuteTB ? ((mB & (Bc - 1)) * Tc + (mB >> 5)) : mB;
    out[(size_t)rA * N + n0 + col] = acc0[v];
    out[(size_t)rB * N + n0 + col] = acc1[v];
  }
}

// ---------------------------------------------------------------------------
// Sequential attention-GRU recurrence. Single workgroup (one WGP), 8 wave32s,
// ~266 KB LDS. Per step:
//   a) q = h @ Wq^T + bq            (WMMA bf16, A = f32 h converted on load)
//   b) scores[b,s] = Wv.tanh(q+kp)  (VALU + shfl reduction, coalesced over H)
//   c) softmax over S, emit attention weights
//   d) ctx = w @ enc ; x = [emb|ctx] packed as bf16 in LDS (aliases q buffer)
//   e) GRU gates in 4 column chunks of 128: r/z (gi+gh summed in-register over
//      K=1024 then K=512), i_n and h_n kept separate; all via WMMA bf16
//   f) pointwise sigmoid/tanh update -> h_new
//   g) commit h, emit h_seq (f32 + bf16 copy for the batched logits GEMM)
// ---------------------------------------------------------------------------
__global__ void decoder_kernel(const float* __restrict__ enc, const float* __restrict__ h0,
                               const float* __restrict__ kp, const __bf16* __restrict__ xembB,
                               const __bf16* __restrict__ WqB, const float* __restrict__ bq,
                               const __bf16* __restrict__ WihB, const __bf16* __restrict__ WhhB,
                               const float* __restrict__ bih, const float* __restrict__ bhh,
                               const float* __restrict__ Wv, const float* __restrict__ bv,
                               float* __restrict__ hseqF, __bf16* __restrict__ hseqB,
                               float* __restrict__ attnOut, float* __restrict__ hFinalOut) {
  extern __shared__ char smem[];
  float*  hS  = (float*)smem;            // [32][512] current hidden (f32)
  float*  hN  = hS + Bc * Hc;            // [32][512] next hidden
  float*  qS  = hN + Bc * Hc;            // [32][512] q   (aliased with xA below)
  __bf16* xA  = (__bf16*)qS;             // [32][1024] x = [emb|ctx] bf16
  float*  gR  = qS + Bc * Hc;            // [32][128] r-gate strip (gi+gh)
  float*  gZ  = gR + Bc * 128;           // [32][128] z-gate strip (gi+gh)
  float*  gIN = gZ + Bc * 128;           // [32][128] i_n strip
  float*  gHN = gIN + Bc * 128;          // [32][128] h_n strip
  float*  scS = gHN + Bc * 128;          // [32][64] scores / attention weights
  float*  wvS = scS + Bc * Sc;           // [512] Wv staged in LDS

  const int tid = threadIdx.x, w = tid >> 5, lane = tid & 31;
  const int col = lane & 15, half = lane >> 4;

  for (int i = tid; i < Bc * Hc; i += 256) hS[i] = h0[i];
  for (int i = tid; i < Hc; i += 256) wvS[i] = Wv[i];
  const float bvv = bv[0];
  __syncthreads();

  for (int t = 0; t < Tc; ++t) {
    // ---- a) q = h @ Wq^T + bq : 64 16x16 tiles, 8 per wave ----------------
    for (int job = w; job < 64; job += 8) {
      const int mt = job & 1, nt = job >> 1, n0 = nt << 4;
      const float bi = bq[n0 + col];
      v8f acc;
#pragma unroll
      for (int v = 0; v < 8; ++v) acc[v] = bi;
      for (int ks = 0; ks < 16; ++ks) {
        v16bf a  = aFragLDSf32(hS, mt << 4, Hc, ks << 5, lane);
        v16bf bb = bFragG(WqB, n0, Hc, ks << 5, lane);
        acc = __builtin_amdgcn_wmma_f32_16x16x32_bf16(false, a, false, bb, (short)0, acc,
                                                      false, false);
      }
#pragma unroll
      for (int v = 0; v < 8; ++v)
        qS[((mt << 4) + v + (half << 3)) * Hc + n0 + col] = acc[v];
    }
    __syncthreads();

    // ---- b) scores[b,s] = Wv . tanh(q[b] + kp[b,s]) + bv ------------------
    for (int p = w << 8; p < (w << 8) + 256; ++p) {  // 2048 (b,s) pairs total
      const int b = p >> 6, s = p & 63;
      const float* kr = kp + ((size_t)((b << 6) + s) << 9);
      const float* qr = qS + (b << 9);
      float acc = 0.f;
      for (int i = 0; i < 16; ++i) {
        const int kk = (i << 5) + lane;  // coalesced across lanes
        acc += wvS[kk] * tanhf(qr[kk] + kr[kk]);
      }
#pragma unroll
      for (int off = 16; off > 0; off >>= 1) acc += __shfl_xor(acc, off);
      if (lane == 0) scS[p] = acc + bvv;
    }
    __syncthreads();

    // ---- c) softmax over S, write attention output ------------------------
    for (int i = 0; i < 4; ++i) {
      const int b = w + (i << 3);
      float v0 = scS[(b << 6) + lane], v1 = scS[(b << 6) + 32 + lane];
      float m = fmaxf(v0, v1);
#pragma unroll
      for (int off = 16; off > 0; off >>= 1) m = fmaxf(m, __shfl_xor(m, off));
      float e0 = expf(v0 - m), e1 = expf(v1 - m);
      float sm = e0 + e1;
#pragma unroll
      for (int off = 16; off > 0; off >>= 1) sm += __shfl_xor(sm, off);
      const float inv = 1.f / sm;
      e0 *= inv; e1 *= inv;
      scS[(b << 6) + lane] = e0;
      scS[(b << 6) + 32 + lane] = e1;
      attnOut[((size_t)b * Tc + t) * Sc + lane] = e0;
      attnOut[((size_t)b * Tc + t) * Sc + 32 + lane] = e1;
    }
    __syncthreads();

    // ---- d) ctx = w @ enc ; pack x = [emb | ctx] as bf16 (overwrites qS) --
    for (int i = 0; i < 64; ++i) {
      const int idx = tid + (i << 8);
      const int b = idx >> 9, k = idx & (Hc - 1);
      const float* er = enc + (((size_t)b << 6) << 9) + k;
      float acc = 0.f;
#pragma unroll 4
      for (int s = 0; s < Sc; ++s) acc += scS[(b << 6) + s] * er[(size_t)s << 9];
      xA[(b << 10) + Hc + k] = (__bf16)acc;
      xA[(b << 10) + k] = xembB[(((size_t)t << 5) + b) * Hc + k];
    }
    __syncthreads();

    // ---- e/f) GRU gates, 4 column chunks of 128 ---------------------------
    for (int jc = 0; jc < Hc; jc += 128) {
      for (int job = w; job < 64; job += 8) {
        const int strip = job >> 4, rem = job & 15, mt = rem & 1, nt = rem >> 1;
        const int nloc = nt << 4;
        int ngBase; float* dstS;
        if      (strip == 0) { ngBase = jc;            dstS = gR;  }  // r: gi+gh
        else if (strip == 1) { ngBase = Hc + jc;       dstS = gZ;  }  // z: gi+gh
        else if (strip == 2) { ngBase = 2 * Hc + jc;   dstS = gIN; }  // i_n only
        else                 { ngBase = 2 * Hc + jc;   dstS = gHN; }  // h_n only
        const int ng = ngBase + nloc;
        const int nW = ng + col;
        float bi;
        if      (strip < 2)  bi = bih[nW] + bhh[nW];
        else if (strip == 2) bi = bih[nW];
        else                 bi = bhh[nW];
        v8f acc;
#pragma unroll
        for (int v = 0; v < 8; ++v) acc[v] = bi;
        if (strip != 3) {  // gi part: x @ W_ih^T, K = 1024
          for (int ks = 0; ks < 32; ++ks) {
            v16bf a  = aFragLDSbf16(xA, mt << 4, 2 * Hc, ks << 5, lane);
            v16bf bb = bFragG(WihB, ng, 2 * Hc, ks << 5, lane);
            acc = __builtin_amdgcn_wmma_f32_16x16x32_bf16(false, a, false, bb, (short)0,
                                                          acc, false, false);
          }
        }
        if (strip != 2) {  // gh part: h @ W_hh^T, K = 512
          for (int ks = 0; ks < 16; ++ks) {
            v16bf a  = aFragLDSf32(hS, mt << 4, Hc, ks << 5, lane);
            v16bf bb = bFragG(WhhB, ng, Hc, ks << 5, lane);
            acc = __builtin_amdgcn_wmma_f32_16x16x32_bf16(false, a, false, bb, (short)0,
                                                          acc, false, false);
          }
        }
#pragma unroll
        for (int v = 0; v < 8; ++v)
          dstS[((mt << 4) + v + (half << 3)) * 128 + nloc + col] = acc[v];
      }
      __syncthreads();

      // pointwise GRU update for this 128-column chunk
      for (int i = 0; i < 16; ++i) {
        const int idx = tid + (i << 8);       // idx = b*128 + jl
        const int b = idx >> 7, jl = idx & 127;
        const float r = 1.f / (1.f + expf(-gR[idx]));
        const float z = 1.f / (1.f + expf(-gZ[idx]));
        const float nn = tanhf(gIN[idx] + r * gHN[idx]);
        const int hj = (b << 9) + jc + jl;
        hN[hj] = (1.f - z) * nn + z * hS[hj];
      }
      __syncthreads();
    }

    // ---- g) commit hidden state -------------------------------------------
    for (int i = tid; i < Bc * Hc; i += 256) {
      const float v = hN[i];
      hseqF[((size_t)t << 14) + i] = v;
      hseqB[((size_t)t << 14) + i] = (__bf16)v;
      hS[i] = v;
      if (t == Tc - 1) hFinalOut[i] = v;
    }
    __syncthreads();
  }
}

// ---------------------------------------------------------------------------
// In-place log-softmax over each row of 32000 (one block per [b,t] row)
// ---------------------------------------------------------------------------
__global__ void log_softmax_rows(float* __restrict__ out) {
  __shared__ float red[8];
  const int tid = threadIdx.x, w = tid >> 5, lane = tid & 31;
  float* row = out + (size_t)blockIdx.x * Vc;

  float m = -3.4e38f;
  for (int i = tid; i < Vc; i += 256) m = fmaxf(m, row[i]);
#pragma unroll
  for (int off = 16; off > 0; off >>= 1) m = fmaxf(m, __shfl_xor(m, off));
  if (lane == 0) red[w] = m;
  __syncthreads();
  float mm = red[0];
#pragma unroll
  for (int i = 1; i < 8; ++i) mm = fmaxf(mm, red[i]);
  __syncthreads();

  float s = 0.f;
  for (int i = tid; i < Vc; i += 256) s += expf(row[i] - mm);
#pragma unroll
  for (int off = 16; off > 0; off >>= 1) s += __shfl_xor(s, off);
  if (lane == 0) red[w] = s;
  __syncthreads();
  float ss = 0.f;
#pragma unroll
  for (int i = 0; i < 8; ++i) ss += red[i];
  const float lse = mm + logf(ss);

  for (int i = tid; i < Vc; i += 256) row[i] = row[i] - lse;
}

// ---------------------------------------------------------------------------
// Host orchestration
// ---------------------------------------------------------------------------
extern "C" void kernel_launch(void* const* d_in, const int* in_sizes, int n_in,
                              void* d_out, int out_size, void* d_ws, size_t ws_size,
                              hipStream_t stream) {
  (void)in_sizes; (void)n_in; (void)out_size; (void)ws_size;

  const float* enc = (const float*)d_in[0];   // [B,S,H]
  const float* ehs = (const float*)d_in[1];   // [1,B,H]
  const int*   tgt = (const int*)d_in[2];     // [B,T]
  const float* emb = (const float*)d_in[3];   // [V,H]
  const float* Wq  = (const float*)d_in[4];
  const float* bq  = (const float*)d_in[5];
  const float* Wk  = (const float*)d_in[6];
  const float* bk  = (const float*)d_in[7];
  const float* Wv  = (const float*)d_in[8];
  const float* bv  = (const float*)d_in[9];
  const float* Wih = (const float*)d_in[10];  // [3H,2H]
  const float* Whh = (const float*)d_in[11];  // [3H,H]
  const float* bih = (const float*)d_in[12];
  const float* bhh = (const float*)d_in[13];
  const float* Wo  = (const float*)d_in[14];  // [V,H]
  const float* bo  = (const float*)d_in[15];
  float* out = (float*)d_out;

  const size_t OUT_H    = (size_t)Bc * Tc * Vc;       // log_probs then h_final
  const size_t OUT_ATTN = OUT_H + (size_t)Bc * Hc;    // then attentions

  // Workspace partition (all 256B aligned; ~53 MB total)
  char* ws = (char*)d_ws;
  size_t off = 0;
  auto take = [&](size_t bytes) -> char* {
    char* p = ws + off;
    off += (bytes + 255) & ~(size_t)255;
    return p;
  };
  __bf16* encB  = (__bf16*)take((size_t)Bc * Sc * Hc * 2);
  __bf16* WkB   = (__bf16*)take((size_t)Hc * Hc * 2);
  __bf16* WqB   = (__bf16*)take((size_t)Hc * Hc * 2);
  __bf16* WihB  = (__bf16*)take((size_t)3 * Hc * 2 * Hc * 2);
  __bf16* WhhB  = (__bf16*)take((size_t)3 * Hc * Hc * 2);
  __bf16* WoB   = (__bf16*)take((size_t)Vc * Hc * 2);
  __bf16* xembB = (__bf16*)take((size_t)Tc * Bc * Hc * 2);
  float*  kproj = (float*)take((size_t)Bc * Sc * Hc * 4);
  float*  hseqF = (float*)take((size_t)Tc * Bc * Hc * 4);
  __bf16* hseqB = (__bf16*)take((size_t)Tc * Bc * Hc * 2);

  auto cvt = [&](const float* s, __bf16* d, int n) {
    cvt_f32_bf16<<<(n + 255) / 256, 256, 0, stream>>>(s, d, n);
  };
  cvt(enc, encB, Bc * Sc * Hc);
  cvt(Wk,  WkB,  Hc * Hc);
  cvt(Wq,  WqB,  Hc * Hc);
  cvt(Wih, WihB, 3 * Hc * 2 * Hc);
  cvt(Whh, WhhB, 3 * Hc * Hc);
  cvt(Wo,  WoB,  Vc * Hc);

  gather_emb<<<(Tc * Bc * Hc) / 256, 256, 0, stream>>>(emb, tgt, xembB);

  // k_proj = enc @ Wk^T + bk : [2048, 512]
  wmma_gemm_bias<<<dim3(Hc / 128, (Bc * Sc) / 32), 256, 0, stream>>>(
      encB, WkB, bk, kproj, Hc, Hc, 0);

  // Sequential recurrence on one WGP (dynamic LDS ~266 KB of the 320 KB pool)
  const size_t smemBytes =
      (size_t)(3 * Bc * Hc + 4 * Bc * 128 + Bc * Sc + Hc) * sizeof(float);
  decoder_kernel<<<1, 256, smemBytes, stream>>>(
      enc, ehs, kproj, xembB, WqB, bq, WihB, WhhB, bih, bhh, Wv, bv,
      hseqF, hseqB, out + OUT_ATTN, out + OUT_H);

  // logits = h_seq @ Wo^T + bo : [2048, 32000], rows permuted [T,B] -> [B,T]
  wmma_gemm_bias<<<dim3(Vc / 128, (Tc * Bc) / 32), 256, 0, stream>>>(
      hseqB, WoB, bo, out, Vc, Hc, 1);

  // log_softmax in place over V
  log_softmax_rows<<<Bc * Tc, 256, 0, stream>>>(out);
}
